// Attention_75703093559813
// MI455X (gfx1250) — matrix operand order repaired and checked by reading, not compiled, not run on hardware
//
#include <hip/hip_runtime.h>

#define MDIM 1024
#define HDIM 1024
#define BATCH 4
#define SEQ 2048
#define BS (BATCH * SEQ)   // 8192 rows
#define NEGV (-1e9f)

typedef _Float16 h16;
typedef __attribute__((ext_vector_type(2)))  __fp16   v2hf;   // cvt_pkrtz result type
typedef __attribute__((ext_vector_type(16))) _Float16 v16h;
typedef __attribute__((ext_vector_type(8)))  float    v8f;
typedef __attribute__((ext_vector_type(4)))  unsigned u32x4;
typedef __attribute__((ext_vector_type(8)))  int      i32x8;
typedef __attribute__((ext_vector_type(4)))  int      i32x4;

union F16Frag { v16h v; unsigned u[8]; };
union CvtU { v2hf h; unsigned u; };

// ---------------------------------------------------------------------------
// TDM 2D tile load: global (row-major, stride_elems f32 per row) -> LDS,
// f32 elements, LDS row padding +2 dwords (row stride = tile_d0 + 2 dwords).
// Descriptor layout per CDNA5 ISA 8.3/8.4 (group0 128b, group1 256b).
// ---------------------------------------------------------------------------
__device__ __forceinline__ void tdm_load_tile_f32(
    unsigned lds_byte_off, const float* gptr,
    unsigned tile_d0, unsigned tile_d1, unsigned stride_elems)
{
    unsigned long long ga = (unsigned long long)(size_t)gptr;
    u32x4 g0;
    g0[0] = 1u;                                   // count=1, user descriptor
    g0[1] = lds_byte_off;                         // lds_addr (bytes)
    g0[2] = (unsigned)ga;                         // global_addr[31:0]
    g0[3] = (unsigned)((ga >> 32) & 0x01FFFFFFull) | (2u << 30);  // addr[56:32] | type=2

    i32x8 g1;
    // data_size=2 (4B) | pad_enable | pad_interval=4 (32 dwords) | pad_amount=1 (2 dwords)
    g1[0] = (int)((2u << 16) | (1u << 20) | (4u << 22) | (1u << 25));
    g1[1] = (int)((tile_d0 & 0xFFFFu) << 16);                     // tensor_dim0 = tile_d0 (lo16)
    g1[2] = (int)((tile_d0 >> 16) | ((tile_d1 & 0xFFFFu) << 16)); // dim0 hi | tensor_dim1 lo
    g1[3] = (int)((tile_d1 >> 16) | (tile_d0 << 16));             // dim1 hi | tile_dim0
    g1[4] = (int)(tile_d1 & 0xFFFFu);                             // tile_dim1 | tile_dim2=0
    g1[5] = (int)stride_elems;                                    // tensor_dim0_stride lo32
    g1[6] = 0;                                                    // stride hi | dim1_stride lo
    g1[7] = 0;
    i32x4 z4 = {0, 0, 0, 0};
    i32x8 z8 = {0, 0, 0, 0, 0, 0, 0, 0};
    __builtin_amdgcn_tensor_load_to_lds(g0, g1, z4, z4, z8, 0);
}

// ---------------------------------------------------------------------------
// Projection GEMM: out[r, h] = sum_m X[r, m] * W[h, m] + bias[h], stored as f16
// Block 256 threads / 8 waves; block tile 64x128; wave tile 32x32 (2x2 WMMA).
// K staged 32/step via double-buffered TDM DMA (f32 in LDS, row stride 34 dw),
// f32->f16 packed converts during fragment build.
// ---------------------------------------------------------------------------
__global__ __launch_bounds__(256) void proj_kernel(
    const float* __restrict__ X, const float* __restrict__ W,
    const float* __restrict__ bias, h16* __restrict__ out)
{
    __shared__ float Xs[2][64 * 34];
    __shared__ float Ws[2][128 * 34];

    const int t    = threadIdx.x;
    const int lane = t & 31;
    const int w    = t >> 5;
    const int lm   = lane & 15;
    const int half = lane >> 4;     // 0/1
    const int wrow = w >> 2;        // 0..1
    const int wcol = w & 3;         // 0..3
    const int r0   = blockIdx.x * 64;
    const int c0   = blockIdx.y * 128;

    // Prologue: stage first K-slab into buffer 0 (one wave issues; EXEC-wide op)
    if (w == 0) {
        tdm_load_tile_f32((unsigned)(size_t)&Xs[0][0], X + (size_t)r0 * MDIM, 32, 64,  MDIM);
        tdm_load_tile_f32((unsigned)(size_t)&Ws[0][0], W + (size_t)c0 * MDIM, 32, 128, MDIM);
    }

    v8f acc[2][2] = {};

    for (int k0 = 0; k0 < MDIM; k0 += 32) {
        const int cur = (k0 >> 5) & 1;
        if (w == 0) {
            if (k0 + 32 < MDIM) {
                // prefetch next slab into the other buffer, then retire current
                tdm_load_tile_f32((unsigned)(size_t)&Xs[cur ^ 1][0],
                                  X + (size_t)r0 * MDIM + k0 + 32, 32, 64,  MDIM);
                tdm_load_tile_f32((unsigned)(size_t)&Ws[cur ^ 1][0],
                                  W + (size_t)c0 * MDIM + k0 + 32, 32, 128, MDIM);
                __builtin_amdgcn_s_wait_tensorcnt(2);   // current buffer's 2 DMAs done
            } else {
                __builtin_amdgcn_s_wait_tensorcnt(0);
            }
        }
        __syncthreads();   // publish staged LDS to all waves

        const float* xs = Xs[cur];
        const float* ws = Ws[cur];

        // A fragments (16x32 f16): lane lm = row M, VGPR j -> K = 16*(j/4)+2*(j%4)+8*half
        F16Frag a[2], bf[2];
        #pragma unroll
        for (int i = 0; i < 2; ++i) {
            int m = 32 * wrow + 16 * i + lm;
            #pragma unroll
            for (int j = 0; j < 8; ++j) {
                int kk = 16 * (j >> 2) + 2 * (j & 3) + 8 * half;
                CvtU cv;
                cv.h = __builtin_amdgcn_cvt_pkrtz(xs[m * 34 + kk], xs[m * 34 + kk + 1]);
                a[i].u[j] = cv.u;
            }
        }
        // B fragments (32x16 f16): lane lm = col N, VGPR j -> K = 16*(j/4)+2*(j%4)+8*half
        #pragma unroll
        for (int j2 = 0; j2 < 2; ++j2) {
            int n = 32 * wcol + 16 * j2 + lm;
            #pragma unroll
            for (int j = 0; j < 8; ++j) {
                int kk = 16 * (j >> 2) + 2 * (j & 3) + 8 * half;
                CvtU cv;
                cv.h = __builtin_amdgcn_cvt_pkrtz(ws[n * 34 + kk], ws[n * 34 + kk + 1]);
                bf[j2].u[j] = cv.u;
            }
        }
        #pragma unroll
        for (int i = 0; i < 2; ++i)
            #pragma unroll
            for (int j2 = 0; j2 < 2; ++j2)
                acc[i][j2] = __builtin_amdgcn_wmma_f32_16x16x32_f16(
                    false, a[i].v, false, bf[j2].v, (short)0, acc[i][j2], false, false);

        __syncthreads();   // all reads done before next TDM overwrites buffer
    }

    // Epilogue: add bias, store f16. C/D layout: reg r -> M = r + 8*half, N = lm
    #pragma unroll
    for (int i = 0; i < 2; ++i) {
        #pragma unroll
        for (int j2 = 0; j2 < 2; ++j2) {
            int col = c0 + 32 * wcol + 16 * j2 + lm;
            float bv = bias[col];
            #pragma unroll
            for (int r = 0; r < 8; ++r) {
                int row = r0 + 32 * wrow + 16 * i + r + 8 * half;
                out[(size_t)row * HDIM + col] = (h16)(acc[i][j2][r] + bv);
            }
        }
    }
}

// ---------------------------------------------------------------------------
// Flash attention: one block per (query-tile of 16, batch). 256 threads / 8 waves.
// Wave w owns m-slice [128w,128w+128) for the QK^T reduction and the same
// 128-wide column slice of the output accumulation. Keys processed 32 per step.
// ---------------------------------------------------------------------------
__global__ __launch_bounds__(256) void attn_kernel(
    const h16* __restrict__ qh, const h16* __restrict__ kh,
    const h16* __restrict__ vh, const int* __restrict__ amask,
    float* __restrict__ out)
{
    __shared__ float sPart[8][16 * 32];   // per-wave partial scores
    __shared__ float sRed[16 * 32];       // reduced + masked scores
    __shared__ h16   Pt[16 * 34];         // exp tile (f16, padded stride)
    __shared__ float rowM[16], rowL[16], rowC[16];

    const int t    = threadIdx.x;
    const int lane = t & 31;
    const int w    = t >> 5;
    const int lm   = lane & 15;
    const int half = lane >> 4;
    const int q0   = blockIdx.x * 16;
    const int b    = blockIdx.y;
    const size_t baseQ = (size_t)(b * SEQ + q0) * HDIM;
    const size_t baseB = (size_t)b * SEQ * HDIM;
    const unsigned short* vhu = (const unsigned short*)vh;

    // Cache q-tile A fragments for this wave's m-slice: 4 chunks of K=32
    F16Frag qa[4];
    #pragma unroll
    for (int c = 0; c < 4; ++c)
        #pragma unroll
        for (int j = 0; j < 8; ++j) {
            int kk = 128 * w + 32 * c + 16 * (j >> 2) + 2 * (j & 3) + 8 * half;
            qa[c].u[j] = *(const unsigned*)(qh + baseQ + (size_t)lm * HDIM + kk);
        }

    v8f oacc[8] = {};                     // 16 x 128 output slice, cols 128w+16*jn
    if (t < 16) { rowM[t] = -3.0e38f; rowL[t] = 0.0f; }
    __syncthreads();

    for (int kc = 0; kc <= q0; kc += 32) {
        // ---- partial scores: two 16x16 key sub-tiles over this wave's m-slice
        #pragma unroll
        for (int kt = 0; kt < 2; ++kt) {
            v8f sacc = {};
            #pragma unroll
            for (int c = 0; c < 4; ++c) {
                F16Frag kb;
                int key = kc + 16 * kt + lm;
                #pragma unroll
                for (int j = 0; j < 8; ++j) {
                    int kk = 128 * w + 32 * c + 16 * (j >> 2) + 2 * (j & 3) + 8 * half;
                    kb.u[j] = (key < SEQ)
                        ? *(const unsigned*)(kh + baseB + (size_t)key * HDIM + kk)
                        : 0u;
                }
                sacc = __builtin_amdgcn_wmma_f32_16x16x32_f16(
                    false, qa[c].v, false, kb.v, (short)0, sacc, false, false);
            }
            #pragma unroll
            for (int r = 0; r < 8; ++r)
                sPart[w][(r + 8 * half) * 32 + 16 * kt + lm] = sacc[r];
        }
        __syncthreads();

        // ---- reduce across waves, scale, mask
        for (int e = t; e < 512; e += 256) {
            int r = e >> 5, cidx = e & 31;
            float s = 0.0f;
            #pragma unroll
            for (int ww = 0; ww < 8; ++ww) s += sPart[ww][e];
            s *= 0.03125f;                          // 1/sqrt(1024)
            int key = kc + cidx;
            bool valid = (key <= q0 + r) && (amask[b * SEQ + key] != 0);
            sRed[e] = valid ? s : NEGV;
        }
        __syncthreads();

        // ---- online-softmax row stats
        if (t < 16) {
            float mold = rowM[t];
            float rmax = -3.0e38f;
            for (int cidx = 0; cidx < 32; ++cidx)
                rmax = fmaxf(rmax, sRed[t * 32 + cidx]);
            float mnew = fmaxf(mold, rmax);
            rowC[t] = __expf(mold - mnew);
            rowM[t] = mnew;
        }
        __syncthreads();

        // ---- P = exp(s - m) as f16
        for (int e = t; e < 512; e += 256) {
            int r = e >> 5, cidx = e & 31;
            Pt[r * 34 + cidx] = (h16)__expf(sRed[e] - rowM[r]);
        }
        __syncthreads();

        if (t < 16) {
            float sum = 0.0f;
            for (int cidx = 0; cidx < 32; ++cidx) sum += (float)Pt[t * 34 + cidx];
            rowL[t] = rowL[t] * rowC[t] + sum;
        }

        // ---- rescale accumulators, then O += P @ V (K=32)
        float cf[8];
        #pragma unroll
        for (int r = 0; r < 8; ++r) cf[r] = rowC[r + 8 * half];
        #pragma unroll
        for (int jn = 0; jn < 8; ++jn)
            #pragma unroll
            for (int r = 0; r < 8; ++r) oacc[jn][r] *= cf[r];

        F16Frag pa;
        #pragma unroll
        for (int j = 0; j < 8; ++j) {
            int kk = 16 * (j >> 2) + 2 * (j & 3) + 8 * half;
            pa.u[j] = *(const unsigned*)&Pt[lm * 34 + kk];
        }
        #pragma unroll
        for (int jn = 0; jn < 8; ++jn) {
            int col = 128 * w + 16 * jn + lm;
            F16Frag vb;
            #pragma unroll
            for (int j = 0; j < 8; ++j) {
                int k1 = kc + 16 * (j >> 2) + 2 * (j & 3) + 8 * half;
                unsigned u0 = (k1     < SEQ) ? (unsigned)vhu[baseB + (size_t)k1       * HDIM + col] : 0u;
                unsigned u1 = (k1 + 1 < SEQ) ? (unsigned)vhu[baseB + (size_t)(k1 + 1) * HDIM + col] : 0u;
                vb.u[j] = u0 | (u1 << 16);
            }
            oacc[jn] = __builtin_amdgcn_wmma_f32_16x16x32_f16(
                false, pa.v, false, vb.v, (short)0, oacc[jn], false, false);
        }
        __syncthreads();
    }

    // ---- normalize and store
    float rl[8];
    #pragma unroll
    for (int r = 0; r < 8; ++r) rl[r] = rowL[r + 8 * half];
    #pragma unroll
    for (int jn = 0; jn < 8; ++jn) {
        int col = 128 * w + 16 * jn + lm;
        #pragma unroll
        for (int r = 0; r < 8; ++r) {
            int row = q0 + r + 8 * half;
            out[((size_t)b * SEQ + row) * HDIM + col] = oacc[jn][r] / rl[r];
        }
    }
}

// ---------------------------------------------------------------------------
extern "C" void kernel_launch(void* const* d_in, const int* in_sizes, int n_in,
                              void* d_out, int out_size, void* d_ws, size_t ws_size,
                              hipStream_t stream)
{
    const float* q     = (const float*)d_in[0];
    const float* k     = (const float*)d_in[1];
    const float* v     = (const float*)d_in[2];
    const int*   amask = (const int*)  d_in[3];
    const float* Wq_w  = (const float*)d_in[4];
    const float* Wq_b  = (const float*)d_in[5];
    const float* Wk_w  = (const float*)d_in[6];
    const float* Wk_b  = (const float*)d_in[7];
    const float* Wv_w  = (const float*)d_in[8];
    const float* Wv_b  = (const float*)d_in[9];

    h16* qh = (h16*)d_ws;
    h16* kh = qh + (size_t)BS * HDIM;
    h16* vh = kh + (size_t)BS * HDIM;

    dim3 pgrid(BS / 64, HDIM / 128);
    proj_kernel<<<pgrid, 256, 0, stream>>>(q, Wq_w, Wq_b, qh);
    proj_kernel<<<pgrid, 256, 0, stream>>>(k, Wk_w, Wk_b, kh);
    proj_kernel<<<pgrid, 256, 0, stream>>>(v, Wv_w, Wv_b, vh);

    attn_kernel<<<dim3(SEQ / 16, BATCH), 256, 0, stream>>>(
        qh, kh, vh, amask, (float*)d_out);
}
